// SelfAttention_CoAtt_42949672961084
// MI455X (gfx1250) — matrix operand order repaired and checked by reading, not compiled, not run on hardware
//
#include <hip/hip_runtime.h>
#include <hip/hip_bf16.h>

// Problem constants
#define Bn   32
#define Ln   512
#define Hn   1024
#define HIDn 512
#define G4n  2048   // 4*HID
#define X2n  2048   // 2*H

typedef __attribute__((ext_vector_type(16))) __bf16 v16bf;
typedef __attribute__((ext_vector_type(8)))  float  v8f;

__device__ __forceinline__ v8f wmma_bf16(v16bf a, v16bf b, v8f c) {
  // v_wmma_f32_16x16x32_bf16: D = A(16x32) * B(32x16) + C(16x16 f32)
  return __builtin_amdgcn_wmma_f32_16x16x32_bf16(false, a, false, b, (short)0, c,
                                                 false, false);
}

// A operand 16x32 bf16, row-major source with stride ld.
// Layout: lanes 0-15 row M=lane, K = 0..15; lanes 16-31 K = 16..31
__device__ __forceinline__ v16bf load_a16x32(const __bf16* __restrict__ src, int ld, int lane) {
  int row = lane & 15;
  int kh  = (lane >> 4) << 4;               // 0 or 16
  const __bf16* p = src + (size_t)row * ld;
#pragma clang diagnostic push
  v16bf a;
#pragma unroll
  for (int j = 0; j < 8; ++j) {
    int k = kh + 2 * j;
    a[2 * j]     = p[k];
    a[2 * j + 1] = p[k + 1];
  }
#pragma clang diagnostic pop
  return a;
}

// B operand 32x16 where B(k,n) = W(ncol0+n, kb+k) and W is row-major (stride ldw).
// k is contiguous in memory -> 16 contiguous bf16 per lane (single 32B load).
// Contiguous k order used here (same as load_a16x32): lanes 0-15 hold K=0..15 of column n, lanes 16-31 hold K=16..31.
__device__ __forceinline__ v16bf load_b_trans(const __bf16* __restrict__ W, int ldw,
                                              int ncol0, int kb, int lane) {
  int nn = lane & 15;
  int kh = (lane >> 4) << 4;                // 0 or 16
  const __bf16* p = W + (size_t)(ncol0 + nn) * ldw + kb + kh;
  v16bf b;
#pragma unroll
  for (int u = 0; u < 16; ++u) b[u] = p[u];
  return b;
}

// B operand 32x16 where B(k,n) = M(kb+k, ncol0+n), M row-major (stride ld): k strided.
__device__ __forceinline__ v16bf load_b_rowmajor(const __bf16* __restrict__ M, int ld,
                                                 int ncol0, int kb, int lane) {
  int nn = lane & 15;
  int kh = (lane >> 4) << 4;
  const __bf16* p = M + (size_t)(kb + kh) * ld + ncol0 + nn;
  v16bf b;
#pragma unroll
  for (int u = 0; u < 16; ++u) b[u] = p[(size_t)u * ld];
  return b;
}

__device__ __forceinline__ float sigmf(float v) { return 1.0f / (1.0f + __expf(-v)); }

// ---------------------------------------------------------------------------
// Kernel 1: f32 -> bf16 conversions, bias fusion, and x-half of new_x
// ---------------------------------------------------------------------------
__global__ void prep_kernel(const float* __restrict__ x,
                            const float* __restrict__ Wih_f, const float* __restrict__ Whh_f,
                            const float* __restrict__ bih_f, const float* __restrict__ bhh_f,
                            const float* __restrict__ Wih_b, const float* __restrict__ Whh_b,
                            const float* __restrict__ bih_b, const float* __restrict__ bhh_b,
                            __bf16* __restrict__ xb, __bf16* __restrict__ newx,
                            __bf16* __restrict__ Wih, __bf16* __restrict__ Whh,
                            float* __restrict__ bias) {
  size_t tid    = (size_t)blockIdx.x * blockDim.x + threadIdx.x;
  size_t stride = (size_t)gridDim.x * blockDim.x;

  const size_t NX = (size_t)Bn * Ln * Hn;
  for (size_t i = tid; i < NX; i += stride) {
    float v = x[i];
    __bf16 bv = (__bf16)v;
    xb[i] = bv;
    size_t b = i / ((size_t)Ln * Hn);
    size_t t = (i / Hn) % Ln;
    size_t h = i % Hn;
    // time-major LSTM input, second half = raw x
    newx[(t * Bn + b) * X2n + Hn + h] = bv;
  }
  const size_t NWI = (size_t)G4n * X2n;
  for (size_t i = tid; i < NWI; i += stride) {
    Wih[i]       = (__bf16)Wih_f[i];
    Wih[NWI + i] = (__bf16)Wih_b[i];
  }
  const size_t NWH = (size_t)G4n * HIDn;
  for (size_t i = tid; i < NWH; i += stride) {
    Whh[i]       = (__bf16)Whh_f[i];
    Whh[NWH + i] = (__bf16)Whh_b[i];
  }
  for (size_t i = tid; i < G4n; i += stride) {
    bias[i]       = bih_f[i] + bhh_f[i];
    bias[G4n + i] = bih_b[i] + bhh_b[i];
  }
}

// ---------------------------------------------------------------------------
// Kernel 2: self-attention. One wave per (batch, 16-row tile).
//   scores(16x512) via WMMA -> LDS, masked softmax, ctx = att @ x via WMMA,
//   ctx written as bf16 into first half of time-major new_x.
// ---------------------------------------------------------------------------
__global__ __launch_bounds__(32) void attn_kernel(const float* __restrict__ amask,
                                                  const __bf16* __restrict__ xb,
                                                  __bf16* __restrict__ newx) {
  const int tile = blockIdx.x;   // 0..31 : 16-row tile of L
  const int b    = blockIdx.y;   // batch
  const int lane = threadIdx.x & 31;

  __shared__ float  s_sc[16][Ln];
  __shared__ __bf16 s_att[16][Ln];

  const __bf16* xbat = xb + (size_t)b * Ln * Hn;
  const __bf16* arow = xbat + (size_t)tile * 16 * Hn;

  const int nn  = lane & 15;
  const int mb8 = (lane >> 4) << 3;

  // ---- scores = x_tile (16xH) @ x^T (Hx512), masked ----
  for (int ct = 0; ct < Ln / 16; ++ct) {
    v8f acc = {};
    for (int ks = 0; ks < Hn / 32; ++ks) {
      v16bf a  = load_a16x32(arow + ks * 32, Hn, lane);
      v16bf bm = load_b_trans(xbat, Hn, ct * 16, ks * 32, lane);  // B(k,j)=x[j][k]
      acc = wmma_bf16(a, bm, acc);
    }
    int j = ct * 16 + nn;
    float msk = amask[(size_t)b * Ln + j];
#pragma unroll
    for (int r = 0; r < 8; ++r) {
      int m = mb8 + r;
      s_sc[m][j] = msk * acc[r] - (1.0f - msk) * 1e20f;
    }
  }
  __syncthreads();

  // ---- softmax per row (lanes 0..15 each own one row) ----
  if (lane < 16) {
    float mx = -3.0e38f;
    for (int j = 0; j < Ln; ++j) mx = fmaxf(mx, s_sc[lane][j]);
    float sum = 0.0f;
    for (int j = 0; j < Ln; ++j) {
      float e = __expf(s_sc[lane][j] - mx);
      s_sc[lane][j] = e;
      sum += e;
    }
    float inv = 1.0f / sum;
    for (int j = 0; j < Ln; ++j) s_att[lane][j] = (__bf16)(s_sc[lane][j] * inv);
  }
  __syncthreads();

  // ---- ctx = att (16x512) @ x (512xH); write into new_x[:, :, 0:H] ----
  for (int dt = 0; dt < Hn / 16; ++dt) {
    v8f acc = {};
    for (int ks = 0; ks < Ln / 32; ++ks) {
      v16bf a  = load_a16x32(&s_att[0][0] + ks * 32, Ln, lane);
      v16bf bm = load_b_rowmajor(xbat, Hn, dt * 16, ks * 32, lane);  // B(k,d)=x[k][d]
      acc = wmma_bf16(a, bm, acc);
    }
#pragma unroll
    for (int r = 0; r < 8; ++r) {
      int i = tile * 16 + mb8 + r;                       // sequence position
      newx[((size_t)i * Bn + b) * X2n + dt * 16 + nn] = (__bf16)acc[r];
    }
  }
}

// ---------------------------------------------------------------------------
// Kernel 3: persistent bidirectional LSTM. 2 workgroups (dir = blockIdx.x),
// 32 waves each. Wave owns 2 fixed (m-tile, h-col-tile) combos: c and h in
// registers for all 512 steps; bf16 h copy in LDS for the WMMA A operand.
// Per combo, 4 gate accumulators (i,f,g,o) over K=2048 (W_ih) + K=512 (W_hh).
// Weights (~10 MB bf16/dir) stay resident in the 192 MB L2.
// ---------------------------------------------------------------------------
__global__ __launch_bounds__(1024) void lstm_kernel(const __bf16* __restrict__ newx,
                                                    const __bf16* __restrict__ Wih,
                                                    const __bf16* __restrict__ Whh,
                                                    const float* __restrict__ bias,
                                                    const int* __restrict__ x_len,
                                                    float* __restrict__ out) {
  const int dir  = blockIdx.x;          // 0 = forward, 1 = backward
  const int tid  = threadIdx.x;
  const int lane = tid & 31;
  const int wave = tid >> 5;

  __shared__ __bf16 s_h[Bn][HIDn];      // 32 KB: current h as bf16 (A operand)

  for (int i = tid; i < Bn * HIDn; i += blockDim.x) (&s_h[0][0])[i] = (__bf16)0.0f;
  __syncthreads();

  const __bf16* W1 = Wih + (size_t)dir * G4n * X2n;
  const __bf16* W2 = Whh + (size_t)dir * G4n * HIDn;
  const float*  bs = bias + (size_t)dir * G4n;

  const int nn  = lane & 15;
  const int mb8 = (lane >> 4) << 3;

  // Wave-owned state tiles
  float creg[2][8], hreg[2][8];
  int   xl[2][8];
  int   mt_[2], j_[2];
#pragma unroll
  for (int cb = 0; cb < 2; ++cb) {
    int combo = wave * 2 + cb;
    mt_[cb] = combo >> 5;               // which 16-row (batch) half
    j_[cb]  = combo & 31;               // which 16-col tile of HID
#pragma unroll
    for (int r = 0; r < 8; ++r) {
      creg[cb][r] = 0.0f;
      hreg[cb][r] = 0.0f;
      xl[cb][r]   = x_len[mt_[cb] * 16 + mb8 + r];
    }
  }

  for (int step = 0; step < Ln; ++step) {
    const int t = dir ? (Ln - 1 - step) : step;

    // prefetch next timestep's x rows while this step's WMMAs run
    if (step + 1 < Ln) {
      int tn = dir ? (t - 1) : (t + 1);
      __builtin_prefetch(newx + ((size_t)tn * Bn) * X2n, 0, 0);
    }

    v8f acc[2][4];
    // ---- phase 1: gate GEMMs (read s_h) ----
#pragma unroll
    for (int cb = 0; cb < 2; ++cb) {
      const int mt = mt_[cb], j = j_[cb];
#pragma unroll
      for (int q = 0; q < 4; ++q) {
        float bv = bs[q * HIDn + j * 16 + nn];
        v8f c0;
#pragma unroll
        for (int r = 0; r < 8; ++r) c0[r] = bv;
        acc[cb][q] = c0;
      }
      const __bf16* arow = newx + ((size_t)t * Bn + mt * 16) * X2n;
      for (int ks = 0; ks < X2n / 32; ++ks) {          // x_t @ W_ih^T, K=2048
        v16bf a = load_a16x32(arow + ks * 32, X2n, lane);
#pragma unroll
        for (int q = 0; q < 4; ++q) {
          v16bf bm = load_b_trans(W1, X2n, q * HIDn + j * 16, ks * 32, lane);
          acc[cb][q] = wmma_bf16(a, bm, acc[cb][q]);
        }
      }
      const __bf16* hrow = &s_h[mt * 16][0];
      for (int ks = 0; ks < HIDn / 32; ++ks) {         // h @ W_hh^T, K=512
        v16bf a = load_a16x32(hrow + ks * 32, HIDn, lane);
#pragma unroll
        for (int q = 0; q < 4; ++q) {
          v16bf bm = load_b_trans(W2, HIDn, q * HIDn + j * 16, ks * 32, lane);
          acc[cb][q] = wmma_bf16(a, bm, acc[cb][q]);
        }
      }
    }
    __syncthreads();   // all reads of s_h complete

    // ---- phase 2: activations + masked state update (write s_h) ----
#pragma unroll
    for (int cb = 0; cb < 2; ++cb) {
      const int mt = mt_[cb], j = j_[cb];
#pragma unroll
      for (int r = 0; r < 8; ++r) {
        float ig = sigmf(acc[cb][0][r]);
        float fg = sigmf(acc[cb][1][r]);
        float gg = tanhf(acc[cb][2][r]);
        float og = sigmf(acc[cb][3][r]);
        float nc = fg * creg[cb][r] + ig * gg;
        float nh = og * tanhf(nc);
        bool valid = (t < xl[cb][r]);
        creg[cb][r] = valid ? nc : creg[cb][r];
        hreg[cb][r] = valid ? nh : hreg[cb][r];
        s_h[mt * 16 + mb8 + r][j * 16 + nn] = (__bf16)hreg[cb][r];
      }
    }
    __syncthreads();   // s_h consistent for next step
  }

  // ---- final h -> out[batch][dir*HID + col] ----
#pragma unroll
  for (int cb = 0; cb < 2; ++cb) {
#pragma unroll
    for (int r = 0; r < 8; ++r) {
      int row = mt_[cb] * 16 + mb8 + r;
      out[(size_t)row * (2 * HIDn) + dir * HIDn + j_[cb] * 16 + nn] = hreg[cb][r];
    }
  }
}

// ---------------------------------------------------------------------------
extern "C" void kernel_launch(void* const* d_in, const int* in_sizes, int n_in,
                              void* d_out, int out_size, void* d_ws, size_t ws_size,
                              hipStream_t stream) {
  const float* x      = (const float*)d_in[0];
  const int*   x_len  = (const int*)d_in[1];
  const float* amask  = (const float*)d_in[2];
  const float* Wih_f  = (const float*)d_in[3];
  const float* Whh_f  = (const float*)d_in[4];
  const float* bih_f  = (const float*)d_in[5];
  const float* bhh_f  = (const float*)d_in[6];
  const float* Wih_b  = (const float*)d_in[7];
  const float* Whh_b  = (const float*)d_in[8];
  const float* bih_b  = (const float*)d_in[9];
  const float* bhh_b  = (const float*)d_in[10];
  float* out = (float*)d_out;

  // Workspace layout (bytes):
  //   [0,   32MB)  x as bf16                (32*512*1024 * 2)
  //   [32,  96MB)  new_x bf16, time-major   (512*32*2048 * 2)
  //   [96, 112MB)  W_ih bf16, both dirs     (2 * 2048*2048 * 2)
  //   [112,116MB)  W_hh bf16, both dirs     (2 * 2048*512 * 2)
  //   [116MB, +16KB) fused bias f32, both dirs
  char* ws = (char*)d_ws;
  __bf16* xb   = (__bf16*)(ws);
  __bf16* newx = (__bf16*)(ws + (32ull << 20));
  __bf16* Wih  = (__bf16*)(ws + (96ull << 20));
  __bf16* Whh  = (__bf16*)(ws + (112ull << 20));
  float*  bias = (float*)(ws + (116ull << 20));

  prep_kernel<<<4096, 256, 0, stream>>>(x, Wih_f, Whh_f, bih_f, bhh_f,
                                        Wih_b, Whh_b, bih_b, bhh_b,
                                        xb, newx, Wih, Whh, bias);

  attn_kernel<<<dim3(Ln / 16, Bn), 32, 0, stream>>>(amask, xb, newx);

  lstm_kernel<<<2, 1024, 0, stream>>>(newx, Wih, Whh, bias, x_len, out);
}
